// MultiHeadAttention_65730179498282
// MI455X (gfx1250) — matrix operand order repaired and checked by reading, not compiled, not run on hardware
//
#include <hip/hip_runtime.h>
#include <hip/hip_bf16.h>
#include <cstdint>

// MI455X (gfx1250) fused multi-head attention, f16 WMMA with f32 accumulate.
// d_ws needs 32 MB: qh/kh/vh ([B,H,T,64] f16, Q pre-scaled) + ctx ([B*T,DIM] f16).

typedef __attribute__((ext_vector_type(16))) _Float16 v16h;
typedef __attribute__((ext_vector_type(8)))  _Float16 v8h;
typedef __attribute__((ext_vector_type(8)))  float    v8f;

#define ATTN_DIM   1024
#define ATTN_H     16
#define ATTN_HD    64
#define ATTN_T     2048
#define ATTN_B     2
#define ATTN_SCALE 0.125f     // HEAD_DIM^-0.5

#define KSTRIDE 48            // 32 k-halves + pad: 96B rows (32B multiple)
#define KVS     80            // 64 halves + pad : 160B rows (32B multiple)
#define PS      72            // 64 halves + pad : 144B rows (16B multiple)

// ---------------- WMMA helpers (contiguous k = 16*(lane>>4) + e order) ----------------

__device__ __forceinline__ v8f wmma_f16(v16h a, v16h b, v8f c) {
  // v_wmma_f32_16x16x32_f16: (neg_a, A, neg_b, B, c_mod, C, reuse_a, reuse_b)
  return __builtin_amdgcn_wmma_f32_16x16x32_f16(false, a, false, b, (short)0, c,
                                                false, false);
}

// A-fragment 16x32 f16: row m = lane&15; element e -> k = (lane>>4)*16 + e
// => one contiguous 16-half run at +((lane>>4)*16), loaded as two 8-half pieces.
__device__ __forceinline__ v16h frag_a(const _Float16* rowbase, int hi) {
  const v8h lo = *(const v8h*)(rowbase + hi * 16);
  const v8h hh = *(const v8h*)(rowbase + hi * 16 + 8);
  v16h r;
#pragma unroll
  for (int i = 0; i < 8; ++i) { r[i] = lo[i]; r[8 + i] = hh[i]; }
  return r;
}

// B-fragment 32x16 f16: col n = lane&15; element e -> k = (lane>>4)*16 + e
// => one contiguous 16-half (32B) run.
__device__ __forceinline__ v16h frag_b(const _Float16* p) {
  return *(const v16h*)p;
}

// Row-wise reductions across the 16 lanes of one C-layout half (wave32;
// xor offsets <= 8 never cross the lane-16 boundary).
__device__ __forceinline__ float half_max16(float v) {
#pragma unroll
  for (int off = 8; off >= 1; off >>= 1) v = fmaxf(v, __shfl_xor(v, off, 32));
  return v;
}
__device__ __forceinline__ float half_sum16(float v) {
#pragma unroll
  for (int off = 8; off >= 1; off >>= 1) v += __shfl_xor(v, off, 32);
  return v;
}

// Async copy 32B (16 halves) global -> LDS, tracked by ASYNCcnt (no VGPR round
// trip). IOFFSET is added to BOTH lds and global addresses (ISA 10, async ops),
// and a flat address truncates to the LDS offset (ISA 10.2 aperture rule).
__device__ __forceinline__ void async_copy_32B(const _Float16* gsrc, _Float16* ldst) {
  uint32_t lds = (uint32_t)(uintptr_t)ldst;
  uint64_t gad = (uint64_t)(uintptr_t)gsrc;
  asm volatile("global_load_async_to_lds_b128 %0, %1, off\n\t"
               "global_load_async_to_lds_b128 %0, %1, off offset:16"
               :: "v"(lds), "v"(gad) : "memory");
}
__device__ __forceinline__ void wait_async_all() {
  asm volatile("s_wait_asynccnt 0x0" ::: "memory");
}

// ---------------- Kernel 1: QKV projection (fp32 in -> f16 head-major) ----------------
// out[m,n] = (X[m,:] . W[n,:]) + bias[n], scattered to [B,H,T,HD] f16, times `scale`.
// Double-buffered LDS: next chunk's global loads overlap current chunk's WMMAs.

__global__ __launch_bounds__(256)
void proj_qkv_kernel(const float* __restrict__ X, const float* __restrict__ W,
                     const float* __restrict__ bias, _Float16* __restrict__ outH,
                     float scale) {
  __shared__ __align__(32) _Float16 As[2][128 * KSTRIDE];
  __shared__ __align__(32) _Float16 Bs[2][128 * KSTRIDE];

  const int tid  = threadIdx.x;
  const int lane = tid & 31;
  const int wav  = tid >> 5;
  const int hi   = lane >> 4;
  const int ln   = lane & 15;
  const int wm   = wav >> 1;             // 4 waves along M (32 rows each)
  const int wn   = wav & 1;              // 2 waves along N (64 cols each)
  const int rowT = blockIdx.y * 128;
  const int colT = blockIdx.x * 128;

  // per-thread staging slots: 4 float4 of X, 4 float4 of W per 128x32 tile
  int srow[4], scol[4];
#pragma unroll
  for (int p = 0; p < 4; ++p) {
    int slot = p * 256 + tid;            // float4 slot 0..1023
    srow[p] = slot >> 3;                 // 8 float4 per 32-wide row
    scol[p] = (slot & 7) * 4;
  }

  v8f acc[2][4] = {};
  float4 xa[4], wb[4];

#pragma unroll
  for (int p = 0; p < 4; ++p) {
    xa[p] = *(const float4*)(X + (size_t)(rowT + srow[p]) * ATTN_DIM + scol[p]);
    wb[p] = *(const float4*)(W + (size_t)(colT + srow[p]) * ATTN_DIM + scol[p]);
  }
#pragma unroll
  for (int p = 0; p < 4; ++p) {
    _Float16* pa = As[0] + srow[p] * KSTRIDE + scol[p];
    _Float16* pb = Bs[0] + srow[p] * KSTRIDE + scol[p];
    pa[0] = (_Float16)xa[p].x; pa[1] = (_Float16)xa[p].y;
    pa[2] = (_Float16)xa[p].z; pa[3] = (_Float16)xa[p].w;
    pb[0] = (_Float16)wb[p].x; pb[1] = (_Float16)wb[p].y;
    pb[2] = (_Float16)wb[p].z; pb[3] = (_Float16)wb[p].w;
  }
  __syncthreads();

  for (int kk = 0; kk < ATTN_DIM; kk += 32) {
    const int  cur  = (kk >> 5) & 1;
    const bool more = (kk + 32) < ATTN_DIM;

    if (more) {                          // stage next chunk into registers now
#pragma unroll
      for (int p = 0; p < 4; ++p) {
        xa[p] = *(const float4*)(X + (size_t)(rowT + srow[p]) * ATTN_DIM + kk + 32 + scol[p]);
        wb[p] = *(const float4*)(W + (size_t)(colT + srow[p]) * ATTN_DIM + kk + 32 + scol[p]);
      }
    }

    v16h a[2], b[4];
#pragma unroll
    for (int mi = 0; mi < 2; ++mi)
      a[mi] = frag_a(As[cur] + (wm * 32 + mi * 16 + ln) * KSTRIDE, hi);
#pragma unroll
    for (int nt = 0; nt < 4; ++nt)
      b[nt] = frag_b(Bs[cur] + (wn * 64 + nt * 16 + ln) * KSTRIDE + hi * 16);
#pragma unroll
    for (int mi = 0; mi < 2; ++mi)
#pragma unroll
      for (int nt = 0; nt < 4; ++nt)
        acc[mi][nt] = wmma_f16(a[mi], b[nt], acc[mi][nt]);

    if (more) {                          // convert + park into the other buffer
#pragma unroll
      for (int p = 0; p < 4; ++p) {
        _Float16* pa = As[cur ^ 1] + srow[p] * KSTRIDE + scol[p];
        _Float16* pb = Bs[cur ^ 1] + srow[p] * KSTRIDE + scol[p];
        pa[0] = (_Float16)xa[p].x; pa[1] = (_Float16)xa[p].y;
        pa[2] = (_Float16)xa[p].z; pa[3] = (_Float16)xa[p].w;
        pb[0] = (_Float16)wb[p].x; pb[1] = (_Float16)wb[p].y;
        pb[2] = (_Float16)wb[p].z; pb[3] = (_Float16)wb[p].w;
      }
    }
    __syncthreads();                     // one barrier per k-step
  }

#pragma unroll
  for (int mi = 0; mi < 2; ++mi) {
#pragma unroll
    for (int nt = 0; nt < 4; ++nt) {
      int colg = colT + wn * 64 + nt * 16 + ln;
      float bv = bias[colg];
      int h = colg >> 6, d = colg & (ATTN_HD - 1);
#pragma unroll
      for (int r = 0; r < 8; ++r) {
        int rowg = rowT + wm * 32 + mi * 16 + r + hi * 8;   // C layout: m=r+8*hi
        int bb = rowg >> 11, t = rowg & (ATTN_T - 1);
        float v = (acc[mi][nt][r] + bv) * scale;
        outH[(((size_t)bb * ATTN_H + h) * ATTN_T + t) * ATTN_HD + d] = (_Float16)v;
      }
    }
  }
}

// ---------------- Kernel 2: flash attention per (b,h), 128 q-rows/block ----------------

__global__ __launch_bounds__(256)
void attn_kernel(const _Float16* __restrict__ Qh, const _Float16* __restrict__ Kh,
                 const _Float16* __restrict__ Vh, _Float16* __restrict__ Ctx) {
  __shared__ __align__(32) _Float16 Ks[64 * KVS];      // K chunk row-major [key][d]
  __shared__ __align__(32) _Float16 Vt[64 * KVS];      // V chunk transposed [d][key]
  __shared__ __align__(32) _Float16 Ps[8][16 * PS];    // per-wave P tile (C->A relayout)

  const int tid  = threadIdx.x;
  const int lane = tid & 31;
  const int wav  = tid >> 5;
  const int hi   = lane >> 4;
  const int ln   = lane & 15;
  const int bh   = blockIdx.y;                          // b*16 + h
  const int qtile = blockIdx.x * 128;

  const size_t headBase = (size_t)bh * ATTN_T * ATTN_HD;

  // Q A-fragments for this wave's 16 rows, pinned in registers (pre-scaled).
  const int qrow = qtile + wav * 16 + ln;
  const _Float16* qp = Qh + headBase + (size_t)qrow * ATTN_HD;
  const v16h aq0 = frag_a(qp + 0,  hi);
  const v16h aq1 = frag_a(qp + 32, hi);

  v8f oacc[4] = {};
  float mrow[8], lrow[8];
#pragma unroll
  for (int r = 0; r < 8; ++r) { mrow[r] = -1e30f; lrow[r] = 0.f; }

  const int cr  = tid >> 2;              // 0..63 key rows (staging)
  const int cc0 = (tid & 3) * 16;        // 0,16,32,48

  for (int kc0 = 0; kc0 < ATTN_T; kc0 += 64) {
    // K chunk: async DMA straight into LDS (ASYNCcnt path, no VGPR round trip)
    async_copy_32B(Kh + headBase + (size_t)(kc0 + cr) * ATTN_HD + cc0,
                   Ks + cr * KVS + cc0);
    // V chunk: must change layout -> load to regs, scatter transposed
    {
      const _Float16* vg = Vh + headBase + (size_t)(kc0 + cr) * ATTN_HD + cc0;
#pragma unroll
      for (int i = 0; i < 16; ++i)
        Vt[(cc0 + i) * KVS + cr] = vg[i];
    }
    if (kc0 + 64 < ATTN_T)               // global_prefetch_b8 for next V chunk
      __builtin_prefetch(Vh + headBase + (size_t)(kc0 + 64 + cr) * ATTN_HD + cc0, 0, 1);
    wait_async_all();                    // s_wait_asynccnt 0 before the barrier
    __syncthreads();

    // S = Q . K^T  (16 q-rows x 64 keys), 8 WMMA
    v8f s[4] = {};
#pragma unroll
    for (int nt = 0; nt < 4; ++nt) {
      const _Float16* kb = Ks + (nt * 16 + ln) * KVS;
      s[nt] = wmma_f16(aq0, frag_b(kb + hi * 16),      s[nt]);
      s[nt] = wmma_f16(aq1, frag_b(kb + 32 + hi * 16), s[nt]);
    }

    // online softmax, P -> per-wave LDS tile (f16)
    _Float16* pw = Ps[wav];
#pragma unroll
    for (int r = 0; r < 8; ++r) {
      float cm = fmaxf(fmaxf(s[0][r], s[1][r]), fmaxf(s[2][r], s[3][r]));
      cm = half_max16(cm);
      float mnew  = fmaxf(mrow[r], cm);
      float alpha = __expf(mrow[r] - mnew);
      mrow[r] = mnew;
      float rs = 0.f;
#pragma unroll
      for (int nt = 0; nt < 4; ++nt) {
        float p = __expf(s[nt][r] - mnew);
        rs += p;
        pw[(r + hi * 8) * PS + nt * 16 + ln] = (_Float16)p;   // C-layout scatter
        oacc[nt][r] *= alpha;
      }
      lrow[r] = lrow[r] * alpha + half_sum16(rs);
    }

    // O += P . V  (16 x 64 += [16x64 keys] . [64 keys x 64 d]), 8 WMMA
#pragma unroll
    for (int kcc = 0; kcc < 64; kcc += 32) {
      v16h pa = frag_a(pw + ln * PS + kcc, hi);
#pragma unroll
      for (int nt = 0; nt < 4; ++nt)
        oacc[nt] = wmma_f16(pa, frag_b(Vt + (nt * 16 + ln) * KVS + kcc + hi * 16),
                            oacc[nt]);
    }
    __syncthreads();
  }

  // normalize, store ctx [B*T, DIM] f16
  const int b = bh >> 4, h = bh & (ATTN_H - 1);
#pragma unroll
  for (int r = 0; r < 8; ++r) {
    float inv = 1.0f / lrow[r];
    int rowg = qtile + wav * 16 + r + hi * 8;
#pragma unroll
    for (int nt = 0; nt < 4; ++nt) {
      int colg = h * ATTN_HD + nt * 16 + ln;
      Ctx[((size_t)b * ATTN_T + rowg) * ATTN_DIM + colg] = (_Float16)(oacc[nt][r] * inv);
    }
  }
}

// ---------------- Kernel 3: output projection (f16 ctx -> fp32 out) ----------------

__global__ __launch_bounds__(256)
void proj_out_kernel(const _Float16* __restrict__ Xh, const float* __restrict__ W,
                     const float* __restrict__ bias, float* __restrict__ Out) {
  __shared__ __align__(32) _Float16 As[2][128 * KSTRIDE];
  __shared__ __align__(32) _Float16 Bs[2][128 * KSTRIDE];

  const int tid  = threadIdx.x;
  const int lane = tid & 31;
  const int wav  = tid >> 5;
  const int hi   = lane >> 4;
  const int ln   = lane & 15;
  const int wm   = wav >> 1;
  const int wn   = wav & 1;
  const int rowT = blockIdx.y * 128;
  const int colT = blockIdx.x * 128;

  int srow[4], scol[4];                 // B staging slots (float4)
#pragma unroll
  for (int p = 0; p < 4; ++p) {
    int slot = p * 256 + tid;
    srow[p] = slot >> 3;
    scol[p] = (slot & 7) * 4;
  }
  int arow[2], acol[2];                 // A staging slots (uint4 = 8 halves)
#pragma unroll
  for (int p = 0; p < 2; ++p) {
    int slot = p * 256 + tid;
    arow[p] = slot >> 2;
    acol[p] = (slot & 3) * 8;
  }

  v8f acc[2][4] = {};
  uint4  av[2];
  float4 wb[4];

#pragma unroll
  for (int p = 0; p < 2; ++p)
    av[p] = *(const uint4*)(Xh + (size_t)(rowT + arow[p]) * ATTN_DIM + acol[p]);
#pragma unroll
  for (int p = 0; p < 4; ++p)
    wb[p] = *(const float4*)(W + (size_t)(colT + srow[p]) * ATTN_DIM + scol[p]);
#pragma unroll
  for (int p = 0; p < 2; ++p)
    *(uint4*)(As[0] + arow[p] * KSTRIDE + acol[p]) = av[p];
#pragma unroll
  for (int p = 0; p < 4; ++p) {
    _Float16* pb = Bs[0] + srow[p] * KSTRIDE + scol[p];
    pb[0] = (_Float16)wb[p].x; pb[1] = (_Float16)wb[p].y;
    pb[2] = (_Float16)wb[p].z; pb[3] = (_Float16)wb[p].w;
  }
  __syncthreads();

  for (int kk = 0; kk < ATTN_DIM; kk += 32) {
    const int  cur  = (kk >> 5) & 1;
    const bool more = (kk + 32) < ATTN_DIM;

    if (more) {
#pragma unroll
      for (int p = 0; p < 2; ++p)
        av[p] = *(const uint4*)(Xh + (size_t)(rowT + arow[p]) * ATTN_DIM + kk + 32 + acol[p]);
#pragma unroll
      for (int p = 0; p < 4; ++p)
        wb[p] = *(const float4*)(W + (size_t)(colT + srow[p]) * ATTN_DIM + kk + 32 + scol[p]);
    }

    v16h a[2], b[4];
#pragma unroll
    for (int mi = 0; mi < 2; ++mi)
      a[mi] = frag_a(As[cur] + (wm * 32 + mi * 16 + ln) * KSTRIDE, hi);
#pragma unroll
    for (int nt = 0; nt < 4; ++nt)
      b[nt] = frag_b(Bs[cur] + (wn * 64 + nt * 16 + ln) * KSTRIDE + hi * 16);
#pragma unroll
    for (int mi = 0; mi < 2; ++mi)
#pragma unroll
      for (int nt = 0; nt < 4; ++nt)
        acc[mi][nt] = wmma_f16(a[mi], b[nt], acc[mi][nt]);

    if (more) {
#pragma unroll
      for (int p = 0; p < 2; ++p)
        *(uint4*)(As[cur ^ 1] + arow[p] * KSTRIDE + acol[p]) = av[p];
#pragma unroll
      for (int p = 0; p < 4; ++p) {
        _Float16* pb = Bs[cur ^ 1] + srow[p] * KSTRIDE + scol[p];
        pb[0] = (_Float16)wb[p].x; pb[1] = (_Float16)wb[p].y;
        pb[2] = (_Float16)wb[p].z; pb[3] = (_Float16)wb[p].w;
      }
    }
    __syncthreads();
  }

#pragma unroll
  for (int mi = 0; mi < 2; ++mi) {
#pragma unroll
    for (int nt = 0; nt < 4; ++nt) {
      int colg = colT + wn * 64 + nt * 16 + ln;
      float bv = bias[colg];
#pragma unroll
      for (int r = 0; r < 8; ++r) {
        int rowg = rowT + wm * 32 + mi * 16 + r + hi * 8;
        Out[(size_t)rowg * ATTN_DIM + colg] = acc[mi][nt][r] + bv;
      }
    }
  }
}

// ---------------- Host launch ----------------

extern "C" void kernel_launch(void* const* d_in, const int* in_sizes, int n_in,
                              void* d_out, int out_size, void* d_ws, size_t ws_size,
                              hipStream_t stream) {
  (void)in_sizes; (void)n_in; (void)out_size; (void)ws_size;
  const float* q  = (const float*)d_in[0];
  const float* k  = (const float*)d_in[1];
  const float* v  = (const float*)d_in[2];
  const float* wq = (const float*)d_in[3];
  const float* bq = (const float*)d_in[4];
  const float* wk = (const float*)d_in[5];
  const float* bk = (const float*)d_in[6];
  const float* wv = (const float*)d_in[7];
  const float* bv = (const float*)d_in[8];
  const float* wo = (const float*)d_in[9];
  const float* bo = (const float*)d_in[10];

  const size_t tok = (size_t)ATTN_B * ATTN_T * ATTN_DIM;   // 4 Mi elements
  _Float16* qh  = (_Float16*)d_ws;
  _Float16* kh  = qh + tok;
  _Float16* vh  = kh + tok;
  _Float16* ctx = vh + tok;

  dim3 blk(256, 1, 1);
  dim3 gProj(ATTN_DIM / 128, (ATTN_B * ATTN_T) / 128, 1);  // (8, 32)
  proj_qkv_kernel<<<gProj, blk, 0, stream>>>(q, wq, bq, qh, ATTN_SCALE);
  proj_qkv_kernel<<<gProj, blk, 0, stream>>>(k, wk, bk, kh, 1.0f);
  proj_qkv_kernel<<<gProj, blk, 0, stream>>>(v, wv, bv, vh, 1.0f);

  dim3 gAttn(ATTN_T / 128, ATTN_B * ATTN_H, 1);            // (16, 32)
  attn_kernel<<<gAttn, blk, 0, stream>>>(qh, kh, vh, ctx);

  proj_out_kernel<<<gProj, blk, 0, stream>>>(ctx, wo, bo, (float*)d_out);
}
